// Indexer_13778255085947
// MI455X (gfx1250) — matrix operand order repaired and checked by reading, not compiled, not run on hardware
//
#include <hip/hip_runtime.h>

// ---------------------------------------------------------------------------
// Types for CDNA5 WMMA (wave32): v16bf A/B operands, v8f f32 accumulator.
// ---------------------------------------------------------------------------
typedef __attribute__((ext_vector_type(16))) __bf16 v16bf;
typedef __attribute__((ext_vector_type(8)))  __bf16 v8bf;
typedef __attribute__((ext_vector_type(4)))  __bf16 v4bf;
typedef __attribute__((ext_vector_type(8)))  float  v8f;
typedef __attribute__((ext_vector_type(4)))  float  v4f;
typedef __attribute__((ext_vector_type(4)))  unsigned int v4u;
typedef __attribute__((ext_vector_type(8)))  int    v8i;
typedef __attribute__((ext_vector_type(4)))  int    v4i;

#if __has_builtin(__builtin_amdgcn_tensor_load_to_lds) && \
    __has_builtin(__builtin_amdgcn_s_wait_tensorcnt)
#define HAVE_TDM 1
#else
#define HAVE_TDM 0
#endif

// Problem constants (fixed by setup_inputs()).
#define T_DIM   16384
#define HID     4096
#define R_DIM   1536
#define H_DIM   32
#define D_DIM   128
#define L_SEQ   4096
#define K_TOP   2048

__device__ __forceinline__ v8f vzero8() { v8f z = {0,0,0,0,0,0,0,0}; return z; }

__device__ __forceinline__ v8bf ld8(const __bf16* p) { return *(const v8bf*)p; }

__device__ __forceinline__ v8bf cvt8(const float* p) {
  v4f a = *(const v4f*)p;
  v4f b = *(const v4f*)(p + 4);
  v8bf r = { (__bf16)a[0], (__bf16)a[1], (__bf16)a[2], (__bf16)a[3],
             (__bf16)b[0], (__bf16)b[1], (__bf16)b[2], (__bf16)b[3] };
  return r;
}

__device__ __forceinline__ v16bf catbf(v8bf lo, v8bf hi) {
  return __builtin_shufflevector(lo, hi, 0,1,2,3,4,5,6,7,8,9,10,11,12,13,14,15);
}

__device__ __forceinline__ v8f wmma_bf16(v16bf a, v16bf b, v8f c) {
  // D = A(16x32 bf16) * B(32x16 bf16) + C(16x16 f32)
  return __builtin_amdgcn_wmma_f32_16x16x32_bf16(false, a, false, b, (short)0, c,
                                                 false, false);
}

// Monotonic float -> uint key (descending float order == descending uint order).
__device__ __forceinline__ unsigned okey(float f) {
  unsigned u = __float_as_uint(f);
  return (u & 0x80000000u) ? ~u : (u | 0x80000000u);
}

// ---------------------------------------------------------------------------
// Prep: fp32 -> bf16 (4-wide).
// ---------------------------------------------------------------------------
__global__ __launch_bounds__(256) void kcvt(const float* __restrict__ s,
                                            void* __restrict__ d_, long n4) {
  v4bf* d = (v4bf*)d_;
  const v4f* sv = (const v4f*)s;
  long i = (long)blockIdx.x * blockDim.x + threadIdx.x;
  long st = (long)gridDim.x * blockDim.x;
  for (; i < n4; i += st) {
    v4f v = sv[i];
    v4bf o = { (__bf16)v[0], (__bf16)v[1], (__bf16)v[2], (__bf16)v[3] };
    d[i] = o;
  }
}

// ---------------------------------------------------------------------------
// Kernel A: per 16-row tile compute [16x4096] @ [4096x160] (bf16 WMMA, f32 acc)
//   cols   0..127 -> LayerNorm -> k_idx (bf16)
//   cols 128..159 -> per-head weights w (fp32)
// LayerNorm stats via __shfl_xor across the 16-lane half-wave (rows live there).
// 8 waves / block, one row-tile per wave.
// ---------------------------------------------------------------------------
__global__ __launch_bounds__(256) void kA_kw(const float* __restrict__ hs,
                                             const void* __restrict__ wkp_,
                                             const float* __restrict__ knw,
                                             const float* __restrict__ knb,
                                             void* __restrict__ kb_,
                                             float* __restrict__ wf) {
  const __bf16* wkp = (const __bf16*)wkp_;   // [160, 4096] bf16 row-major
  __bf16* kb = (__bf16*)kb_;                 // [T, 128] bf16
  const int tid  = threadIdx.x;
  const int lane = tid & 31;
  const int t0   = (blockIdx.x * 8 + (tid >> 5)) * 16;

  const int rowA = lane & 15;
  const int kA   = (lane >> 4) * 8;    // A K-half base
  const int colB = lane & 15;
  const int kB   = (lane >> 4) * 8;    // B K-half base
  const int hi8  = (lane >> 4) * 8;    // C/D row-group base

  float wgt[8], bia[8];
#pragma unroll
  for (int n = 0; n < 8; ++n) {
    int d = n * 16 + colB;
    wgt[n] = knw[d];
    bia[n] = knb[d];
  }

  v8f acc[10];
#pragma unroll
  for (int n = 0; n < 10; ++n) acc[n] = vzero8();

  const float* arow = hs + (size_t)(t0 + rowA) * HID;
  for (int kc = 0; kc < HID / 32; ++kc) {
    const int kofs = kc * 32 + kA;
    v16bf a = catbf(cvt8(arow + kofs), cvt8(arow + kofs + 16));
#pragma unroll
    for (int n = 0; n < 10; ++n) {
      const __bf16* bp = wkp + (size_t)(n * 16 + colB) * HID + kc * 32 + kB;
      v16bf b = catbf(ld8(bp), ld8(bp + 16));
      acc[n] = wmma_bf16(a, b, acc[n]);
    }
  }

  // LayerNorm stats for the 8 rows this half-wave owns.
  float mu[8], rs[8];
#pragma unroll
  for (int v = 0; v < 8; ++v) {
    float s = 0.f, q = 0.f;
#pragma unroll
    for (int n = 0; n < 8; ++n) { float x = acc[n][v]; s += x; q += x * x; }
#pragma unroll
    for (int m = 1; m < 16; m <<= 1) {
      s += __shfl_xor(s, m, 16);
      q += __shfl_xor(q, m, 16);
    }
    float m_  = s * (1.0f / 128.0f);
    float var = q * (1.0f / 128.0f) - m_ * m_;
    mu[v] = m_;
    rs[v] = rsqrtf(var + 1e-6f);
  }

#pragma unroll
  for (int n = 0; n < 8; ++n)
#pragma unroll
    for (int v = 0; v < 8; ++v) {
      float o = (acc[n][v] - mu[v]) * rs[v] * wgt[n] + bia[n];
      kb[(size_t)(t0 + hi8 + v) * D_DIM + n * 16 + colB] = (__bf16)o;
    }
#pragma unroll
  for (int n = 8; n < 10; ++n)
#pragma unroll
    for (int v = 0; v < 8; ++v)
      wf[(size_t)(t0 + hi8 + v) * H_DIM + (n - 8) * 16 + colB] = acc[n][v];
}

// ---------------------------------------------------------------------------
// Kernel B: q_comb[t,:] = sum_h w[t,h] * (q_latent[t,:] @ wq_b[h*128:(h+1)*128,:].T)
// wq_b tiles staged in LDS (48KB: 128 rows x 192 bf16 = 6 K-chunks), shared by
// all 8 waves. On CDNA5 the stage is a Tensor Data Mover DMA: wave 0 issues
// TENSOR_LOAD_TO_LDS with a D# describing a 128x384B tile (row stride 3072B),
// waits on TENSORcnt, then the block barriers. Fallback: cooperative copy.
// ---------------------------------------------------------------------------
#define BROWS 128      // rows per staged tile (one head's d-block)
#define BKC   6        // K-chunks (of 32) per stage
#define BROWE (BKC*32) // 192 bf16 per row in LDS

__global__ __launch_bounds__(256) void kB_qcomb(const void* __restrict__ qb_,
                                                const void* __restrict__ wqbb_,
                                                const float* __restrict__ wf,
                                                void* __restrict__ qcb_) {
  const __bf16* qb   = (const __bf16*)qb_;     // [T, 1536]
  const __bf16* wqbb = (const __bf16*)wqbb_;   // [4096, 1536]
  __bf16* qcb = (__bf16*)qcb_;                 // [T, 128]
  __shared__ __bf16 Bs[BROWS * BROWE];         // 48 KB

  const int tid  = threadIdx.x;
  const int lane = tid & 31;
  const int wave = tid >> 5;
  const int t0   = (blockIdx.x * 8 + wave) * 16;

  const int rowA = lane & 15;
  const int kA   = (lane >> 4) * 8;
  const int colB = lane & 15;
  const int kB   = (lane >> 4) * 8;
  const int hi8  = (lane >> 4) * 8;

  const __bf16* arow = qb + (size_t)(t0 + rowA) * R_DIM;

  v8f qacc[8];
#pragma unroll
  for (int n = 0; n < 8; ++n) qacc[n] = vzero8();

  for (int h = 0; h < H_DIM; ++h) {
    v8f hacc[8];
#pragma unroll
    for (int n = 0; n < 8; ++n) hacc[n] = vzero8();

    for (int st = 0; st < (R_DIM / 32) / BKC; ++st) {
      __syncthreads();  // previous stage fully consumed
#if HAVE_TDM
      if (wave == 0) {
        const unsigned long long ga =
            (unsigned long long)(size_t)(wqbb + (size_t)(h * BROWS) * R_DIM +
                                         st * BROWE);
        const unsigned ldsoff = (unsigned)(size_t)(&Bs[0]);
        // D# group0: count=1 | lds_addr | global_addr[56:0] | type=2
        v4u g0 = { 1u, ldsoff, (unsigned)(ga & 0xFFFFFFFFu),
                   (unsigned)((ga >> 32) & 0x01FFFFFFu) | (2u << 30) };
        // D# group1 (data_size=8B units): tensor_dim0=384 (3072B row),
        // tensor_dim1=128 rows, tile_dim0=48 (384B), tile_dim1=128,
        // tensor_dim0_stride=384.
        v8i g1 = { (int)(3u << 16),        // wg_mask=0 | data_size=3 (8B)
                   (int)(384u << 16),      // tensor_dim0[15:0] << 16
                   (int)(128u << 16),      // dim0 hi=0 | tensor_dim1[15:0]
                   (int)(48u << 16),       // dim1 hi=0 | tile_dim0=48
                   (int)128,               // tile_dim1=128 | tile_dim2=0
                   (int)384,               // tensor_dim0_stride[31:0]
                   0, 0 };                 // stride hi, dim1_stride
        v4i z4 = {0, 0, 0, 0};             // groups 2/3: tensor <= 2D
        v8i z8 = {0, 0, 0, 0, 0, 0, 0, 0}; // trailing descriptor group (unused)
        __builtin_amdgcn_tensor_load_to_lds(g0, g1, z4, z4, z8, 0);
        __builtin_amdgcn_s_wait_tensorcnt(0);
      }
      __syncthreads();
#else
      {
        // 256 threads x 192B: thread copies half a row (96 elems, 6x32B).
        const __bf16* src = wqbb + (size_t)(h * BROWS + (tid >> 1)) * R_DIM +
                            st * BROWE + (tid & 1) * 96;
        __bf16* dst = Bs + (tid >> 1) * BROWE + (tid & 1) * 96;
#pragma unroll
        for (int q = 0; q < 6; ++q)
          *(v16bf*)(dst + q * 16) = *(const v16bf*)(src + q * 16);
      }
      __syncthreads();
#endif
#pragma unroll
      for (int s = 0; s < BKC; ++s) {
        const int kc = st * BKC + s;
        v16bf a = catbf(ld8(arow + kc * 32 + kA), ld8(arow + kc * 32 + kA + 16));
#pragma unroll
        for (int n = 0; n < 8; ++n) {
          const __bf16* bp = Bs + (n * 16 + colB) * BROWE + s * 32 + kB;
          v16bf b = catbf(ld8(bp), ld8(bp + 16));
          hacc[n] = wmma_bf16(a, b, hacc[n]);
        }
      }
    }

    float wv[8];
#pragma unroll
    for (int v = 0; v < 8; ++v) wv[v] = wf[(size_t)(t0 + hi8 + v) * H_DIM + h];
#pragma unroll
    for (int n = 0; n < 8; ++n)
#pragma unroll
      for (int v = 0; v < 8; ++v) qacc[n][v] += wv[v] * hacc[n][v];
  }

#pragma unroll
  for (int n = 0; n < 8; ++n)
#pragma unroll
    for (int v = 0; v < 8; ++v)
      qcb[(size_t)(t0 + hi8 + v) * D_DIM + n * 16 + colB] = (__bf16)qacc[n][v];
}

// ---------------------------------------------------------------------------
// Kernel C: scores (WMMA, causal mask) into 256KB LDS, then per-row
// 4x8-bit radix select of the 2048th largest + deterministic ballot compaction.
// One block (8 waves) per 16-row query tile.
// ---------------------------------------------------------------------------
#define SC_ELEMS (16 * L_SEQ)

__global__ __launch_bounds__(256) void kC_topk(const void* __restrict__ qcb_,
                                               const void* __restrict__ kb_,
                                               int* __restrict__ out) {
  extern __shared__ char smem[];
  float* sc  = (float*)smem;                      // [16][4096] scores
  int* hist  = (int*)(smem + SC_ELEMS * 4);       // 256 bins
  int* ctl   = hist + 256;                        // ctl[0]=prefix ctl[1]=krem
  int* wsg   = ctl + 8;                           // per-wave >thresh counts
  int* wse   = ctl + 16;                          // per-wave ==thresh counts

  const __bf16* qcb = (const __bf16*)qcb_;
  const __bf16* kb  = (const __bf16*)kb_;

  const int tid  = threadIdx.x;
  const int lane = tid & 31;
  const int wave = tid >> 5;
  const int t0   = blockIdx.x * 16;
  const int seq_start = t0 & ~(L_SEQ - 1);
  const int jt_count  = ((t0 - seq_start) >> 4) + 1;

  for (int i = tid; i < SC_ELEMS; i += 256) sc[i] = -__builtin_inff();
  __syncthreads();

  const int rowA = lane & 15;
  const int kA   = (lane >> 4) * 8;
  const int colB = lane & 15;
  const int kB   = (lane >> 4) * 8;
  const int hi8  = (lane >> 4) * 8;

  // A (q_comb tile) held in registers for all j-tiles.
  const __bf16* arow = qcb + (size_t)(t0 + rowA) * D_DIM;
  v16bf a[4];
#pragma unroll
  for (int kc = 0; kc < 4; ++kc)
    a[kc] = catbf(ld8(arow + kc * 32 + kA), ld8(arow + kc * 32 + kA + 16));

  for (int jt = wave; jt < jt_count; jt += 8) {
    const int jg0 = seq_start + jt * 16;
    const __bf16* brow = kb + (size_t)(jg0 + colB) * D_DIM;
    if (jt + 8 < jt_count)
      __builtin_prefetch(kb + (size_t)(jg0 + 128 + colB) * D_DIM, 0, 1);

    v8f acc = vzero8();
#pragma unroll
    for (int kc = 0; kc < 4; ++kc) {
      v16bf b = catbf(ld8(brow + kc * 32 + kB), ld8(brow + kc * 32 + kB + 16));
      acc = wmma_bf16(a[kc], b, acc);
    }
    const int jg = jg0 + colB;
#pragma unroll
    for (int v = 0; v < 8; ++v) {
      const int ig = t0 + hi8 + v;
      float s = (jg > ig) ? -__builtin_inff() : acc[v];
      sc[(hi8 + v) * L_SEQ + jt * 16 + colB] = s;
    }
  }
  __syncthreads();

  for (int row = 0; row < 16; ++row) {
    const float* rp = sc + row * L_SEQ;

    // Radix select: find key of the K_TOP-th largest.
    unsigned prefix = 0;
    int krem = K_TOP;
    for (int shift = 24; shift >= 0; shift -= 8) {
      hist[tid] = 0;
      __syncthreads();
      for (int s = 0; s < L_SEQ / 256; ++s) {
        const int j = tid + s * 256;
        const unsigned key = okey(rp[j]);
        const bool in =
            (shift == 24) || (((key ^ prefix) >> (shift + 8)) == 0u);
        if (in) atomicAdd(&hist[(key >> shift) & 255], 1);
      }
      __syncthreads();
      if (tid == 0) {
        int kr = krem;
        int b = 255;
        for (; b > 0; --b) {
          int ccount = hist[b];
          if (ccount >= kr) break;
          kr -= ccount;
        }
        ctl[0] = (int)(prefix | ((unsigned)b << shift));
        ctl[1] = kr;
      }
      __syncthreads();
      prefix = (unsigned)ctl[0];
      krem   = ctl[1];
    }
    const unsigned thresh = prefix;
    const int nties = krem;            // ties at threshold to keep
    const int ngt   = K_TOP - nties;   // strictly-greater count

    // Deterministic compaction: j-ascending within each class.
    int* orow = out + (size_t)(t0 + row) * K_TOP;
    int run_gt = 0, run_eq = 0;
    for (int ch = 0; ch < L_SEQ / 256; ++ch) {
      const int j = ch * 256 + tid;
      const unsigned key = okey(rp[j]);
      const bool fg = key > thresh;
      const bool fe = key == thresh;
      const unsigned long long bg = __ballot(fg);
      const unsigned long long be = __ballot(fe);
      if (lane == 0) { wsg[wave] = __popcll(bg); wse[wave] = __popcll(be); }
      __syncthreads();
      int bgt = run_gt, beq = run_eq, tg = 0, te = 0;
#pragma unroll
      for (int w = 0; w < 8; ++w) {
        if (w < wave) { bgt += wsg[w]; beq += wse[w]; }
        tg += wsg[w];
        te += wse[w];
      }
      const unsigned long long lm = ((unsigned long long)1 << lane) - 1;
      if (fg) {
        int p = bgt + __popcll(bg & lm);
        if (p < K_TOP) orow[p] = seq_start + j;
      }
      if (fe) {
        int p = beq + __popcll(be & lm);
        if (p < nties) orow[ngt + p] = seq_start + j;
      }
      run_gt += tg;
      run_eq += te;
      __syncthreads();
    }
  }
}

// ---------------------------------------------------------------------------
// Host launcher.
// ---------------------------------------------------------------------------
extern "C" void kernel_launch(void* const* d_in, const int* in_sizes, int n_in,
                              void* d_out, int out_size, void* d_ws,
                              size_t ws_size, hipStream_t stream) {
  (void)in_sizes; (void)n_in; (void)out_size; (void)ws_size;
  const float* hs    = (const float*)d_in[0];  // [1,T,HID]
  const float* ql    = (const float*)d_in[1];  // [1,T,R]
  const float* wqb   = (const float*)d_in[2];  // [H*D,R]
  const float* wk    = (const float*)d_in[3];  // [D,HID]
  const float* knw   = (const float*)d_in[4];  // [D]
  const float* knb   = (const float*)d_in[5];  // [D]
  const float* wproj = (const float*)d_in[6];  // [H,HID]
  int* out = (int*)d_out;                      // [1,T,1,K] int32 indices

  // Workspace carve (bf16 staging + intermediates), all offsets 256B aligned.
  char* ws = (char*)d_ws;
  size_t off = 0;
  void* qb   = ws + off; off += (size_t)T_DIM * R_DIM * 2;      // 48 MB
  void* wqbb = ws + off; off += (size_t)(H_DIM * D_DIM) * R_DIM * 2; // 12 MB
  char* wkpb = ws + off; off += (size_t)(D_DIM + H_DIM) * HID * 2;   // 1.25 MB
  void* kbuf = ws + off; off += (size_t)T_DIM * D_DIM * 2;      // 4 MB
  float* wf  = (float*)(ws + off); off += (size_t)T_DIM * H_DIM * 4; // 2 MB
  void* qcb  = ws + off; off += (size_t)T_DIM * D_DIM * 2;      // 4 MB

  // 1) Prep: bf16 conversions for reused GEMM operands.
  kcvt<<<2048, 256, 0, stream>>>(ql, qb, (long)T_DIM * R_DIM / 4);
  kcvt<<<1024, 256, 0, stream>>>(wqb, wqbb, (long)(H_DIM * D_DIM) * R_DIM / 4);
  kcvt<<<256, 256, 0, stream>>>(wk, (void*)wkpb, (long)D_DIM * HID / 4);
  kcvt<<<64, 256, 0, stream>>>(wproj, (void*)(wkpb + (size_t)D_DIM * HID * 2),
                               (long)H_DIM * HID / 4);

  // 2) Fused k_idx (+LayerNorm) and per-head weights.
  kA_kw<<<T_DIM / (16 * 8), 256, 0, stream>>>(hs, (const void*)wkpb, knw, knb,
                                              kbuf, wf);

  // 3) Head-weighted q combination (dominant GEMM, TDM-staged B tiles).
  kB_qcomb<<<T_DIM / (16 * 8), 256, 0, stream>>>(qb, wqbb, wf, qcb);

  // 4) Causal scores + per-row top-2048 radix select.
  const int smem_bytes = SC_ELEMS * 4 + 256 * 4 + 24 * 4 + 64;
  kC_topk<<<T_DIM / 16, 256, smem_bytes, stream>>>(qcb, kbuf, out);
}
